// GCN_15066745274644
// MI455X (gfx1250) — compile-verified, matrix operand order checked by reading, not run on hardware
//
#include <hip/hip_runtime.h>
#include <hip/hip_bf16.h>
#include <stdint.h>
#include <stddef.h>

// ---------------------------------------------------------------------------
// GCN 2-layer forward for MI455X (gfx1250, wave32).
//   * all heavy GEMMs: v_wmma_f32_16x16x32_bf16, fp32 accumulate
//   * adjacency GEMMs: deterministic split-K (S=8) for occupancy + register-
//     prefetch software pipeline over the K tiles
//   * cosine GEMM: double-buffered LDS fed by global_load_async_to_lds_b128
//     (ASYNCcnt-pipelined, s_wait_asynccnt 4 retires exactly one tile)
// Workspace use: ~310 MB (two 8192^2 bf16 matrices + split-K partials).
// ---------------------------------------------------------------------------

typedef __attribute__((ext_vector_type(16))) __bf16 v16bf;
typedef __attribute__((ext_vector_type(8)))  float  v8f;

#define LEAKY_SLOPE 0.01f
#define COS_EPS 1e-8f
#define NROW 8192
#define DFEAT 128
#define LDS_STRIDE 48   // bf16 elems per LDS row: 96 B, multiple of 16 B
#define SPLITK 8

__device__ __forceinline__ unsigned short f2bf(float f) {
  unsigned u = __float_as_uint(f);
  u += 0x7fffu + ((u >> 16) & 1u);          // RNE (NaN not expected here)
  return (unsigned short)(u >> 16);
}
__device__ __forceinline__ float bf2f(unsigned short h) {
  return __uint_as_float(((unsigned)h) << 16);
}

union FragBF { v16bf v; uint4 q[2]; };

// CDNA5 async copy: 16 B/lane global -> LDS, tracked by ASYNCcnt.
__device__ __forceinline__ void async_copy_b128(unsigned lds_addr, const void* g) {
  asm volatile("global_load_async_to_lds_b128 %0, %1, off"
               :: "v"(lds_addr), "v"((unsigned long long)(uintptr_t)g)
               : "memory");
}

// ---- fp32 -> packed bf16, 4 elems / thread ---------------------------------
__global__ void k_f32_to_bf16(const float* __restrict__ src,
                              unsigned short* __restrict__ dst, int n4) {
  int i = blockIdx.x * blockDim.x + threadIdx.x;
  if (i >= n4) return;
  float4 f = ((const float4*)src)[i];
  uint2 o;
  o.x = (unsigned)f2bf(f.x) | ((unsigned)f2bf(f.y) << 16);
  o.y = (unsigned)f2bf(f.z) | ((unsigned)f2bf(f.w) << 16);
  ((uint2*)dst)[i] = o;
}

// ---- row L2 norms of a [rows x 128] bf16 matrix ----------------------------
__global__ void k_rownorm(const unsigned short* __restrict__ X,
                          float* __restrict__ nrm, int rows) {
  int r = blockIdx.x * blockDim.x + threadIdx.x;
  if (r >= rows) return;
  const unsigned short* p = X + (size_t)r * DFEAT;
  float s = 0.f;
  for (int i = 0; i < DFEAT; i += 2) {
    unsigned u = *(const unsigned*)&p[i];
    float a = bf2f((unsigned short)u);
    float b = bf2f((unsigned short)(u >> 16));
    s += a * a + b * b;
  }
  nrm[r] = sqrtf(s);
}

// ---- split-K reduce + optional leaky-ReLU + fp32->bf16 ---------------------
__global__ void k_reduce_act(const float* __restrict__ part, int S,
                             unsigned short* __restrict__ out, int n4, int act) {
  int i = blockIdx.x * blockDim.x + threadIdx.x;
  if (i >= n4) return;
  float4 s = make_float4(0.f, 0.f, 0.f, 0.f);
  for (int j = 0; j < S; ++j) {               // fixed order -> deterministic
    float4 p = ((const float4*)(part + (size_t)j * NROW * DFEAT))[i];
    s.x += p.x; s.y += p.y; s.z += p.z; s.w += p.w;
  }
  if (act) {
    s.x = (s.x >= 0.f) ? s.x : LEAKY_SLOPE * s.x;
    s.y = (s.y >= 0.f) ? s.y : LEAKY_SLOPE * s.y;
    s.z = (s.z >= 0.f) ? s.z : LEAKY_SLOPE * s.z;
    s.w = (s.w >= 0.f) ? s.w : LEAKY_SLOPE * s.w;
  }
  uint2 o;
  o.x = (unsigned)f2bf(s.x) | ((unsigned)f2bf(s.y) << 16);
  o.y = (unsigned)f2bf(s.z) | ((unsigned)f2bf(s.w) << 16);
  ((uint2*)out)[i] = o;
}

// ---- split-K bf16 WMMA GEMM, N fixed at 128 --------------------------------
// Cpart[blockIdx.y] [M x 128] fp32 = A[m-tile, ks:ke] @ B[ks:ke, :]
// A: [M x K] bf16 row-major (lda); transA=1 reads A as [K x M].
// Register-prefetch pipeline: global loads of tile k+32 overlap WMMAs of k.
__global__ __launch_bounds__(256)
void k_gemm_splitk(const unsigned short* __restrict__ A, long lda, int transA,
                   const unsigned short* __restrict__ B,
                   float* __restrict__ Cpart, int K, int kChunk) {
  __shared__ __align__(16) unsigned short At[128 * LDS_STRIDE];
  __shared__ __align__(16) unsigned short Bt[128 * LDS_STRIDE];

  const int tid  = threadIdx.x;
  const int lane = tid & 31;
  const int wave = tid >> 5;
  const int m0   = blockIdx.x * 128;
  const int lr   = lane & 15;
  const int kb   = (lane >> 4) * 8;
  const int ks   = blockIdx.y * kChunk;
  const int ke   = (ks + kChunk < K) ? ks + kChunk : K;

  float* __restrict__ Cp = Cpart + (size_t)blockIdx.y * NROW * DFEAT;

  v8f acc[8] = {};

  // per-thread staging chunks: 2 for A-tile, 2 for B-tile
  const int ia0 = tid, ia1 = tid + 256;        // in [0,512)
  uint4 rA0, rA1, rB0, rB1;

  auto fetchA = [&](int k0, int idx) -> uint4 {
    if (!transA) {
      int m = idx >> 2, kc = (idx & 3) << 3;
      return *(const uint4*)&A[(size_t)(m0 + m) * lda + k0 + kc];
    } else {
      int k = idx >> 4, mc = (idx & 15) << 3;
      return *(const uint4*)&A[(size_t)(k0 + k) * lda + m0 + mc];
    }
  };
  auto fetchB = [&](int k0, int idx) -> uint4 {
    int k = idx >> 4, nc = (idx & 15) << 3;
    return *(const uint4*)&B[(size_t)(k0 + k) * DFEAT + nc];
  };
  auto storeA = [&](int idx, uint4 d) {
    if (!transA) {
      int m = idx >> 2, kc = (idx & 3) << 3;
      *(uint4*)&At[m * LDS_STRIDE + kc] = d;
    } else {
      int k = idx >> 4, mc = (idx & 15) << 3;
      const unsigned short* e = (const unsigned short*)&d;
#pragma unroll
      for (int j = 0; j < 8; ++j) At[(mc + j) * LDS_STRIDE + k] = e[j];
    }
  };
  auto storeB = [&](int idx, uint4 d) {   // Bt[n][k] = B[k][n]
    int k = idx >> 4, nc = (idx & 15) << 3;
    const unsigned short* e = (const unsigned short*)&d;
#pragma unroll
    for (int j = 0; j < 8; ++j) Bt[(nc + j) * LDS_STRIDE + k] = e[j];
  };

  // prologue fetch
  rA0 = fetchA(ks, ia0); rA1 = fetchA(ks, ia1);
  rB0 = fetchB(ks, ia0); rB1 = fetchB(ks, ia1);

  for (int k0 = ks; k0 < ke; k0 += 32) {
    __syncthreads();                 // previous compute done before overwrite
    storeA(ia0, rA0); storeA(ia1, rA1);
    storeB(ia0, rB0); storeB(ia1, rB1);
    __syncthreads();

    if (k0 + 32 < ke) {              // overlap next-tile fetch with WMMAs
      rA0 = fetchA(k0 + 32, ia0); rA1 = fetchA(k0 + 32, ia1);
      rB0 = fetchB(k0 + 32, ia0); rB1 = fetchB(k0 + 32, ia1);
    }

    FragBF a;
    const int arow = wave * 16 + lr;
    a.q[0] = *(const uint4*)&At[arow * LDS_STRIDE + kb];
    a.q[1] = *(const uint4*)&At[arow * LDS_STRIDE + kb + 16];

    FragBF b[8];
#pragma unroll
    for (int ns = 0; ns < 8; ++ns) {  // preload all B frags, then WMMA burst
      const int bcol = ns * 16 + lr;
      b[ns].q[0] = *(const uint4*)&Bt[bcol * LDS_STRIDE + kb];
      b[ns].q[1] = *(const uint4*)&Bt[bcol * LDS_STRIDE + kb + 16];
    }
#pragma unroll
    for (int ns = 0; ns < 8; ++ns)
      acc[ns] = __builtin_amdgcn_wmma_f32_16x16x32_bf16(
          false, a.v, false, b[ns].v, (short)0, acc[ns], false, false);
  }

  // fp32 partial store: m = m0 + wave*16 + r + (lane>=16 ? 8 : 0), n = ns*16+lr
  const int mbase = m0 + wave * 16 + (lane >> 4) * 8;
#pragma unroll
  for (int ns = 0; ns < 8; ++ns) {
    const int n = ns * 16 + lr;
#pragma unroll
    for (int r = 0; r < 8; ++r)
      Cp[(size_t)(mbase + r) * DFEAT + n] = acc[ns][r];
  }
}

// ---- cosine-similarity GEMM with async-to-LDS double buffering -------------
// out[m][n] = (P[m]·Q[n]) / max(nP[m]*nQ[n], eps);  P,Q: [8192 x 128] bf16.
__global__ __launch_bounds__(256)
void k_gemm_cos(const unsigned short* __restrict__ P,
                const unsigned short* __restrict__ Q,
                const float* __restrict__ nP, const float* __restrict__ nQ,
                unsigned short* __restrict__ outB, float* __restrict__ outF) {
  __shared__ __align__(16) unsigned short At[2][128 * LDS_STRIDE];
  __shared__ __align__(16) unsigned short Bt[2][128 * LDS_STRIDE];

  const int tid  = threadIdx.x;
  const int lane = tid & 31;
  const int wave = tid >> 5;
  const int m0   = blockIdx.x * 128;
  const int n0   = blockIdx.y * 128;
  const int lr   = lane & 15;
  const int kb   = (lane >> 4) * 8;

  v8f acc[8] = {};

  // issue one K-tile (32 wide) of async copies: 1024 x 16B chunks,
  // 4 chunks/thread -> ASYNCcnt += 4 per wave.
  auto issue_tile = [&](int k0, int bufi) {
    for (int idx = tid; idx < 1024; idx += 256) {
      unsigned lds; const unsigned short* g;
      if (idx < 512) {
        int m = idx >> 2, kc = (idx & 3) << 3;
        lds = (unsigned)(uintptr_t)&At[bufi][m * LDS_STRIDE + kc];
        g   = &P[(size_t)(m0 + m) * DFEAT + k0 + kc];
      } else {
        int j = idx - 512;
        int n = j >> 2, kc = (j & 3) << 3;
        lds = (unsigned)(uintptr_t)&Bt[bufi][n * LDS_STRIDE + kc];
        g   = &Q[(size_t)(n0 + n) * DFEAT + k0 + kc];
      }
      async_copy_b128(lds, g);
    }
  };

  issue_tile(0, 0);
  for (int t = 0; t < 4; ++t) {                 // K = 128 = 4 tiles of 32
    if (t < 3) {
      issue_tile((t + 1) * 32, (t + 1) & 1);
      asm volatile("s_wait_asynccnt 0x4" ::: "memory");  // tile t landed
    } else {
      asm volatile("s_wait_asynccnt 0x0" ::: "memory");
    }
    __syncthreads();

    const unsigned short* at = At[t & 1];
    const unsigned short* bt = Bt[t & 1];
    FragBF a;
    const int arow = wave * 16 + lr;
    a.q[0] = *(const uint4*)&at[arow * LDS_STRIDE + kb];
    a.q[1] = *(const uint4*)&at[arow * LDS_STRIDE + kb + 16];
    FragBF b[8];
#pragma unroll
    for (int ns = 0; ns < 8; ++ns) {
      const int bcol = ns * 16 + lr;
      b[ns].q[0] = *(const uint4*)&bt[bcol * LDS_STRIDE + kb];
      b[ns].q[1] = *(const uint4*)&bt[bcol * LDS_STRIDE + kb + 16];
    }
#pragma unroll
    for (int ns = 0; ns < 8; ++ns)
      acc[ns] = __builtin_amdgcn_wmma_f32_16x16x32_bf16(
          false, a.v, false, b[ns].v, (short)0, acc[ns], false, false);
    __syncthreads();                 // compute done before buffer reuse (t+2)
  }

  const int mbase = m0 + wave * 16 + (lane >> 4) * 8;
#pragma unroll
  for (int ns = 0; ns < 8; ++ns) {
    const int n = n0 + ns * 16 + lr;
    const float qn = nQ[n];
#pragma unroll
    for (int r = 0; r < 8; ++r) {
      const int m = mbase + r;
      float denom = fmaxf(nP[m] * qn, COS_EPS);
      float v = acc[ns][r] / denom;
      if (outF) outF[(size_t)m * NROW + n] = v;
      else      outB[(size_t)m * NROW + n] = f2bf(v);
    }
  }
}

// ---------------------------------------------------------------------------
extern "C" void kernel_launch(void* const* d_in, const int* in_sizes, int n_in,
                              void* d_out, int out_size, void* d_ws, size_t ws_size,
                              hipStream_t stream) {
  const float* pre = (const float*)d_in[0];   // [8192,128]
  const float* cur = (const float*)d_in[1];   // [8192,128]
  const float* adj = (const float*)d_in[2];   // [8192,8192]
  const float* W1  = (const float*)d_in[3];   // [128,128]
  const float* W2  = (const float*)d_in[4];   // [128,128]
  float* out = (float*)d_out;                 // [8192,8192]

  char* ws = (char*)d_ws;
  size_t off = 0;
  auto alloc = [&](size_t bytes) {
    void* p = ws + off;
    off = (off + bytes + 255) & ~(size_t)255;
    return p;
  };
  const size_t ADJ_E = (size_t)NROW * NROW;
  const size_t FEA_E = (size_t)NROW * DFEAT;
  unsigned short* adjb  = (unsigned short*)alloc(ADJ_E * 2);
  unsigned short* adj1b = (unsigned short*)alloc(ADJ_E * 2);
  unsigned short* preb  = (unsigned short*)alloc(FEA_E * 2);
  unsigned short* curb  = (unsigned short*)alloc(FEA_E * 2);
  unsigned short* w1b   = (unsigned short*)alloc((size_t)DFEAT * DFEAT * 2);
  unsigned short* w2b   = (unsigned short*)alloc((size_t)DFEAT * DFEAT * 2);
  unsigned short* prew  = (unsigned short*)alloc(FEA_E * 2);
  unsigned short* curw  = (unsigned short*)alloc(FEA_E * 2);
  unsigned short* pre1  = (unsigned short*)alloc(FEA_E * 2);
  unsigned short* cur1  = (unsigned short*)alloc(FEA_E * 2);
  unsigned short* pre2w = (unsigned short*)alloc(FEA_E * 2);
  unsigned short* cur2w = (unsigned short*)alloc(FEA_E * 2);
  unsigned short* pre2  = (unsigned short*)alloc(FEA_E * 2);
  unsigned short* cur2  = (unsigned short*)alloc(FEA_E * 2);
  float* n1p = (float*)alloc(NROW * 4);
  float* n1c = (float*)alloc(NROW * 4);
  float* n2p = (float*)alloc(NROW * 4);
  float* n2c = (float*)alloc(NROW * 4);
  float* cpart = (float*)alloc((size_t)SPLITK * FEA_E * 4);   // 32 MB
  (void)ws_size; (void)in_sizes; (void)n_in; (void)out_size;

  auto cvt = [&](const float* s, unsigned short* d, size_t n) {
    int n4 = (int)(n / 4);
    k_f32_to_bf16<<<(n4 + 255) / 256, 256, 0, stream>>>(s, d, n4);
  };

  const dim3 blk(256);
  const int gM = NROW / 128;                       // 64
  const dim3 gCos(gM, gM);
  const int nOut4 = (int)(FEA_E / 4);              // 256Ki
  const int gRed = (nOut4 + 255) / 256;

  // GEMM helper: Out[Mx128] = act(A @ B), split-K S ways, fp32 partials.
  auto gemm = [&](const unsigned short* A, long lda, int transA,
                  const unsigned short* B, unsigned short* Out,
                  int K, int S, int act) {
    int kChunk = K / S;
    k_gemm_splitk<<<dim3(gM, S), blk, 0, stream>>>(A, lda, transA, B,
                                                   cpart, K, kChunk);
    k_reduce_act<<<gRed, blk, 0, stream>>>(cpart, S, Out, nOut4, act);
  };

  // ---- bf16 conversions ----
  cvt(adj, adjb, ADJ_E);
  cvt(pre, preb, FEA_E);
  cvt(cur, curb, FEA_E);
  cvt(W1, w1b, (size_t)DFEAT * DFEAT);
  cvt(W2, w2b, (size_t)DFEAT * DFEAT);

  // ---- layer 1 ----
  gemm(preb, DFEAT, 0, w1b, prew, DFEAT, 1, 0);
  gemm(curb, DFEAT, 0, w1b, curw, DFEAT, 1, 0);
  gemm(adjb, NROW, 0, curw, pre1, NROW, SPLITK, 1);
  gemm(adjb, NROW, 1, prew, cur1, NROW, SPLITK, 1);
  k_rownorm<<<NROW / 256, blk, 0, stream>>>(pre1, n1p, NROW);
  k_rownorm<<<NROW / 256, blk, 0, stream>>>(cur1, n1c, NROW);
  k_gemm_cos<<<gCos, blk, 0, stream>>>(pre1, cur1, n1p, n1c, adj1b, nullptr);

  // ---- layer 2 ----
  gemm(pre1, DFEAT, 0, w2b, pre2w, DFEAT, 1, 0);
  gemm(cur1, DFEAT, 0, w2b, cur2w, DFEAT, 1, 0);
  gemm(adj1b, NROW, 0, cur2w, pre2, NROW, SPLITK, 1);
  gemm(adj1b, NROW, 1, pre2w, cur2, NROW, SPLITK, 1);
  k_rownorm<<<NROW / 256, blk, 0, stream>>>(pre2, n2p, NROW);
  k_rownorm<<<NROW / 256, blk, 0, stream>>>(cur2, n2c, NROW);
  k_gemm_cos<<<gCos, blk, 0, stream>>>(pre2, cur2, n2p, n2c, nullptr, out);
}
